// PKM_78941498901188
// MI455X (gfx1250) — compile-verified, matrix operand order checked by reading, not run on hardware
//
#include <hip/hip_runtime.h>

typedef __attribute__((ext_vector_type(16))) _Float16 v16h;
typedef __attribute__((ext_vector_type(8)))  _Float16 v8h;
typedef __attribute__((ext_vector_type(4)))  _Float16 v4h;
typedef __attribute__((ext_vector_type(8)))  float    v8f;
typedef unsigned int u32x4 __attribute__((ext_vector_type(4)));
typedef int          i32x8 __attribute__((ext_vector_type(8)));
typedef int          i32x4 __attribute__((ext_vector_type(4)));

#define NB    4
#define NT    2048
#define NDIM  1024
#define NH    8
#define NDH   128
#define NKEY  256
#define NTOP  32
#define NROWS (NB * NT)          // 8192
#define NZ    16                 // (p,h) batches

#if defined(__gfx1250__) && __has_builtin(__builtin_amdgcn_tensor_load_to_lds) && \
    __has_builtin(__builtin_amdgcn_s_wait_tensorcnt)
#define USE_TDM 1
#else
#define USE_TDM 0
#endif

// ---------------------------------------------------------------- conversions
__global__ __launch_bounds__(256) void cvt_f32_f16(const float* __restrict__ in,
                                                   _Float16* __restrict__ out, int n) {
  int i = blockIdx.x * 256 + threadIdx.x;
  if (i < n) out[i] = (_Float16)in[i];
}

// keys[h][n][p][d] (f32) -> keys16[z=(p*8+h)][n][d] (f16)
__global__ __launch_bounds__(256) void permute_keys(const float* __restrict__ keys,
                                                    _Float16* __restrict__ out) {
  int idx = blockIdx.x * 256 + threadIdx.x;       // 8*256*2*128 = 524288
  int d = idx & 127;
  int p = (idx >> 7) & 1;
  int n = (idx >> 8) & 255;
  int h = idx >> 16;
  out[((size_t)(p * 8 + h) * 256 + n) * 128 + d] = (_Float16)keys[idx];
}

#if USE_TDM
// TDM: 2-D tile load global->LDS. tile_x elems (data_size=2B) x tile_y rows,
// row stride = stride_elems. LDS padding: +16B after every 64B row so the
// LDS image has the 40-half padded stride the fragment reader expects.
// Descriptor packing per CDNA5 ISA section 8.3/8.4 (groups 2/3 zero => 2-D).
__device__ inline void tdm_load_2d_b16(unsigned lds_off, const void* gptr,
                                       unsigned tile_x, unsigned tile_y,
                                       unsigned stride_elems) {
  unsigned long long ga = (unsigned long long)(size_t)gptr;
  u32x4 g0;
  g0[0] = 1u;                                          // count=1 (valid user D#)
  g0[1] = lds_off;                                     // lds_addr
  g0[2] = (unsigned)ga;                                // global_addr[31:0]
  g0[3] = (unsigned)((ga >> 32) & 0x1FFFFFFu) | (2u << 30);  // addr[56:32] | type=2
  i32x8 g1;
  g1[0] = (int)((1u << 16)        // data_size = 1 -> 2 bytes
                | (1u << 20)      // pad_enable
                | (3u << 22)      // pad_interval: 16 DWORDs (64B)
                | (3u << 25));    // pad_amount: 4 DWORDs (16B)
  g1[1] = (int)((tile_x & 0xFFFFu) << 16);                               // tensor_dim0 lo
  g1[2] = (int)(((tile_x >> 16) & 0xFFFFu) | ((tile_y & 0xFFFFu) << 16)); // td0 hi | td1 lo
  g1[3] = (int)(((tile_y >> 16) & 0xFFFFu) | ((tile_x & 0xFFFFu) << 16)); // td1 hi | tile_dim0
  g1[4] = (int)(tile_y & 0xFFFFu);                                        // tile_dim1 (dim2=0)
  g1[5] = (int)stride_elems;                                              // dim0_stride lo32
  g1[6] = 0;                                                              // stride hi | d1s lo
  g1[7] = 0;
  i32x4 z4 = {0, 0, 0, 0};
  i32x8 z8 = {0, 0, 0, 0, 0, 0, 0, 0};
  __builtin_amdgcn_tensor_load_to_lds(g0, g1, z4, z4, z8, 0);
}
#endif

// ---------------------------------------------------------------- WMMA GEMM
// C[M,N] = A[M,K] (row-major f16) * Bt[N,K]^T (row-major f16), f32 accum.
// Workgroup tile 64x128, 8 waves (2x4), each wave 32x32 (2x2 WMMA tiles).
// LDS tiles stored M|N x K row-major, stride 40 halves (pad, 16B aligned).
// Fragment per ISA 16-bit layout: lane L: row = L%16, kb = (L<16?0:8);
// halves [kb,kb+8) and [16+kb,16+kb+8)  -> two ds_load_b128.
__device__ inline v16h load_frag(const _Float16* base, int kb) {
  v8h lo = *(const v8h*)(base + kb);
  v8h hi = *(const v8h*)(base + 16 + kb);
  v16h r;
#pragma unroll
  for (int i = 0; i < 8; ++i) { r[i] = lo[i]; r[i + 8] = hi[i]; }
  return r;
}

template <int MODE>   // 0: f16 out   1: f32 out   2: f32 + bias out
__global__ __launch_bounds__(256) void gemm_wmma(
    const _Float16* __restrict__ A,  long long strideA, int lda,
    const _Float16* __restrict__ Bt, long long strideB, int ldb,
    void* __restrict__ Cv,           long long strideC, int ldc,
    int K, const float* __restrict__ bias)
{
  __shared__ __align__(16) _Float16 lA[64 * 40];
  __shared__ __align__(16) _Float16 lB[128 * 40];

  const int tid   = threadIdx.x;
  const int lane  = tid & 31;
  const int wave  = tid >> 5;
  const int waveM = wave >> 2;             // 0..1
  const int waveN = wave & 3;              // 0..3
  const int batch = blockIdx.z;
  const int m0 = blockIdx.y * 64;
  const int n0 = blockIdx.x * 128;

  A  += (size_t)batch * strideA;
  Bt += (size_t)batch * strideB;

  const int rA = tid >> 2, cA = (tid & 3) * 8;
#if USE_TDM
  const unsigned lB_off = (unsigned)(size_t)(void*)&lB[0];
#endif

  v8f acc[2][2];
#pragma unroll
  for (int i = 0; i < 2; ++i)
#pragma unroll
    for (int j = 0; j < 2; ++j)
      acc[i][j] = (v8f){0.f, 0.f, 0.f, 0.f, 0.f, 0.f, 0.f, 0.f};

  for (int k0 = 0; k0 < K; k0 += 32) {
    // issue global loads into registers early (overlap with barrier)
    v8h ta = *(const v8h*)&A[(size_t)(m0 + rA) * lda + k0 + cA];
#if !USE_TDM
    v8h tb0 = *(const v8h*)&Bt[(size_t)(n0 + rA) * ldb + k0 + cA];
    v8h tb1 = *(const v8h*)&Bt[(size_t)(n0 + rA + 64) * ldb + k0 + cA];
#endif
    if (k0 + 32 < K)
      __builtin_prefetch(&A[(size_t)(m0 + rA) * lda + k0 + 32 + cA], 0, 1);

    __syncthreads();                 // previous iteration's LDS reads complete

#if USE_TDM
    if (wave == 0)                   // one DMA per workgroup (EXEC ignored)
      tdm_load_2d_b16(lB_off, &Bt[(size_t)n0 * ldb + k0], 32u, 128u, (unsigned)ldb);
#endif
    *(v8h*)&lA[rA * 40 + cA] = ta;   // A tile 64x32, one b128 per thread
#if !USE_TDM
    *(v8h*)&lB[rA * 40 + cA] = tb0;  // B tile 128x32, two b128 per thread
    *(v8h*)&lB[(rA + 64) * 40 + cA] = tb1;
#endif
#if USE_TDM
    __builtin_amdgcn_s_wait_tensorcnt(0);   // DMA complete (no-op for waves 1..7)
#endif
    __syncthreads();                 // staged tiles visible to all waves

    const int kb = (lane >> 4) * 8;
    const int fr = lane & 15;
    v16h af[2], bf[2];
#pragma unroll
    for (int mi = 0; mi < 2; ++mi)
      af[mi] = load_frag(&lA[(waveM * 32 + mi * 16 + fr) * 40], kb);
#pragma unroll
    for (int ni = 0; ni < 2; ++ni)
      bf[ni] = load_frag(&lB[(waveN * 32 + ni * 16 + fr) * 40], kb);

#pragma unroll
    for (int mi = 0; mi < 2; ++mi)
#pragma unroll
      for (int ni = 0; ni < 2; ++ni)
        acc[mi][ni] = __builtin_amdgcn_wmma_f32_16x16x32_f16(
            false, af[mi], false, bf[ni], (short)0, acc[mi][ni], false, false);
  }

  // epilogue — C/D layout: lane: N = L%16, rows r + (L>=16 ? 8 : 0)
  const int cn    = lane & 15;
  const int rbase = (lane >> 4) * 8;
  _Float16* C16 = (MODE == 0) ? ((_Float16*)Cv + (size_t)batch * strideC) : nullptr;
  float*    C32 = (MODE != 0) ? ((float*)Cv    + (size_t)batch * strideC) : nullptr;
#pragma unroll
  for (int mi = 0; mi < 2; ++mi)
#pragma unroll
    for (int ni = 0; ni < 2; ++ni) {
      int gc = n0 + waveN * 32 + ni * 16 + cn;
#pragma unroll
      for (int r = 0; r < 8; ++r) {
        int gr = m0 + waveM * 32 + mi * 16 + rbase + r;
        float vv = acc[mi][ni][r];
        if (MODE == 0) {
          C16[(size_t)gr * ldc + gc] = (_Float16)vv;
        } else {
          if (MODE == 2) vv += bias[gc];
          C32[(size_t)gr * ldc + gc] = vv;
        }
      }
    }
}

// ------------------------------------------------- top-32 of 256, one wave/row
__global__ __launch_bounds__(256) void topk_256_32(const float* __restrict__ dots,
                                                   float* __restrict__ scores,
                                                   int* __restrict__ sidx)
{
  const int gw   = (blockIdx.x * 256 + threadIdx.x) >> 5;  // row 0..131071
  const int lane = threadIdx.x & 31;
  const float* row = dots + (size_t)gw * 256;

  float v[8]; int id[8];
  float4 a4 = *(const float4*)(row + lane * 8);
  float4 b4 = *(const float4*)(row + lane * 8 + 4);
  v[0] = a4.x; v[1] = a4.y; v[2] = a4.z; v[3] = a4.w;
  v[4] = b4.x; v[5] = b4.y; v[6] = b4.z; v[7] = b4.w;
#pragma unroll
  for (int j = 0; j < 8; ++j) id[j] = lane * 8 + j;

  // Batcher network, descending, ties -> smaller index first
#define CE(i, j)                                                              \
  { if (v[i] < v[j] || (v[i] == v[j] && id[i] > id[j])) {                     \
      float tv = v[i]; v[i] = v[j]; v[j] = tv;                                \
      int ti = id[i]; id[i] = id[j]; id[j] = ti; } }
  CE(0,1) CE(2,3) CE(4,5) CE(6,7)
  CE(0,2) CE(1,3) CE(4,6) CE(5,7)
  CE(1,2) CE(5,6)
  CE(0,4) CE(1,5) CE(2,6) CE(3,7)
  CE(2,4) CE(3,5)
  CE(1,2) CE(3,4) CE(5,6)
#undef CE

  float* so = scores + (size_t)gw * 32;
  int*   io = sidx   + (size_t)gw * 32;
#pragma unroll 1
  for (int k = 0; k < 32; ++k) {
    float cv = v[0]; int ci = id[0];
#pragma unroll
    for (int off = 16; off > 0; off >>= 1) {
      float ov = __shfl_xor(cv, off);
      int   oi = __shfl_xor(ci, off);
      if (ov > cv || (ov == cv && oi < ci)) { cv = ov; ci = oi; }
    }
    if (lane == k) { so[k] = cv; io[k] = ci; }
    if ((ci >> 3) == lane) {       // winner's owner: pop head of sorted list
      v[0]=v[1]; id[0]=id[1]; v[1]=v[2]; id[1]=id[2]; v[2]=v[3]; id[2]=id[3];
      v[3]=v[4]; id[3]=id[4]; v[4]=v[5]; id[4]=id[5]; v[5]=v[6]; id[5]=id[6];
      v[6]=v[7]; id[6]=id[7]; v[7] = -3.0e38f; id[7] = 0x7fffffff;
    }
  }
}

// ------- combine halves, top-32 of 32x32 sum grid, softmax, gather values ----
// one wave per (b,h,t'):  gw = ((b*8+h)*2048 + t'),  t' = 2*i + p
__global__ __launch_bounds__(256) void combine_gather(
    const float* __restrict__ scores, const int* __restrict__ sidx,
    const float* __restrict__ values, _Float16* __restrict__ ctx16)
{
  const int gw   = (blockIdx.x * 256 + threadIdx.x) >> 5;  // 0..65535
  const int lane = threadIdx.x & 31;
  const int tp = gw & 2047;
  const int h  = (gw >> 11) & 7;
  const int b  = gw >> 14;
  const int i  = tp >> 1, p = tp & 1;
  const int z  = p * 8 + h;

  const size_t r0 = (size_t)z * NROWS + (size_t)b * NT + i;          // s0 row
  const size_t r1 = r0 + 1024;                                       // s1 row
  float a  = scores[r0 * 32 + lane];  int ia = sidx[r0 * 32 + lane]; // lane=ki
  float bs = scores[r1 * 32 + lane];  int ib = sidx[r1 * 32 + lane]; // lane=kj

  // frontier merge: lane ki proposes a + s1[ptr], sorted descending in ptr
  int ptr = 0;
  float selv = -3.0e38f; int selc = 0;
#pragma unroll 1
  for (int k = 0; k < 32; ++k) {
    float s1v = __shfl(bs, ptr & 31);
    float cv  = (ptr < 32) ? (a + s1v) : -3.0e38f;
    int   ci  = (lane << 5) | (ptr & 31);
#pragma unroll
    for (int off = 16; off > 0; off >>= 1) {
      float ov = __shfl_xor(cv, off);
      int   oi = __shfl_xor(ci, off);
      if (ov > cv || (ov == cv && oi < ci)) { cv = ov; ci = oi; }
    }
    if (lane == k) { selv = cv; selc = ci; }
    if ((ci >> 5) == lane) ptr++;
  }

  // value index = i0[ki]*256 + i1[kj]
  int i0s  = __shfl(ia, selc >> 5);
  int i1s  = __shfl(ib, selc & 31);
  int vidx = i0s * 256 + i1s;

  // softmax over the 32 lane-held selections
  float m = selv;
#pragma unroll
  for (int off = 16; off > 0; off >>= 1) m = fmaxf(m, __shfl_xor(m, off));
  float e = __expf(selv - m);
  float s = e;
#pragma unroll
  for (int off = 16; off > 0; off >>= 1) s += __shfl_xor(s, off);
  float w = e / s;

  // gather 32 rows of 128 f32 from values[h], weighted sum (lane = 4 floats)
  const float* vh = values + (size_t)h * (NKEY * NKEY) * NDH;
  float ax = 0.f, ay = 0.f, az = 0.f, aw = 0.f;
#pragma unroll 1
  for (int k = 0; k < 32; ++k) {
    float wk = __shfl(w, k);
    int   vk = __shfl(vidx, k);
    float4 vv = *(const float4*)(vh + (size_t)vk * NDH + lane * 4);
    ax += wk * vv.x; ay += wk * vv.y; az += wk * vv.z; aw += wk * vv.w;
  }

  // ctx16[b, t', h*128 + lane*4 .. +4]  (f16, feeds output WMMA GEMM)
  v4h o; o[0] = (_Float16)ax; o[1] = (_Float16)ay; o[2] = (_Float16)az; o[3] = (_Float16)aw;
  *(v4h*)(ctx16 + (size_t)(b * NT + tp) * NDIM + h * NDH + lane * 4) = o;
}

// ---------------------------------------------------------------- launcher
extern "C" void kernel_launch(void* const* d_in, const int* in_sizes, int n_in,
                              void* d_out, int out_size, void* d_ws, size_t ws_size,
                              hipStream_t stream) {
  (void)in_sizes; (void)n_in; (void)out_size; (void)ws_size;
  const float* x      = (const float*)d_in[0];   // (4,2048,1024)
  const float* wq     = (const float*)d_in[1];   // (2048,1024)
  const float* wout   = (const float*)d_in[2];   // (1024,1024)
  const float* bias   = (const float*)d_in[3];   // (1024,)
  const float* keys   = (const float*)d_in[4];   // (8,256,2,128)
  const float* values = (const float*)d_in[5];   // (8,65536,128)
  float* out = (float*)d_out;                    // (4,2048,1024) f32

  char* ws = (char*)d_ws;
  _Float16* x16    = (_Float16*)(ws);                         // 16 MB
  _Float16* wq16   = (_Float16*)(ws + (16u << 20));           //  4 MB
  _Float16* wout16 = (_Float16*)(ws + (20u << 20));           //  2 MB
  _Float16* keys16 = (_Float16*)(ws + (22u << 20));           //  1 MB
  _Float16* q16    = (_Float16*)(ws + (23u << 20));           // 32 MB  (z,row,128)
  _Float16* ctx16  = (_Float16*)(ws + (55u << 20));           // 16 MB  (row,1024)
  float*    dots   = (float*)   (ws + (71u << 20));           // 128 MB (z,row,256)
  float*    scores = (float*)   (ws + (199u << 20));          // 16 MB  (z,row,32)
  int*      sidx   = (int*)     (ws + (215u << 20));          // 16 MB

  const int nx = NROWS * NDIM, nwq = 2 * NDIM * NDIM, nwo = NDIM * NDIM;
  cvt_f32_f16<<<(nx  + 255) / 256, 256, 0, stream>>>(x, x16, nx);
  cvt_f32_f16<<<(nwq + 255) / 256, 256, 0, stream>>>(wq, wq16, nwq);
  cvt_f32_f16<<<(nwo + 255) / 256, 256, 0, stream>>>(wout, wout16, nwo);
  permute_keys<<<(NH * NKEY * 2 * NDH) / 256, 256, 0, stream>>>(keys, keys16);

  // 1) q16[z][r][d] = sum_e x[r][e] * Wq[z*128+d][e]     (batched over z)
  gemm_wmma<0><<<dim3(1, NROWS / 64, NZ), 256, 0, stream>>>(
      x16, 0LL, NDIM, wq16, (long long)NDH * NDIM, NDIM,
      (void*)q16, (long long)NROWS * NDH, NDH, NDIM, nullptr);

  // 2) dots[z][r][n] = sum_d q16[z][r][d] * keys16[z][n][d]
  gemm_wmma<1><<<dim3(NKEY / 128, NROWS / 64, NZ), 256, 0, stream>>>(
      q16, (long long)NROWS * NDH, NDH, keys16, (long long)NKEY * NDH, NDH,
      (void*)dots, (long long)NROWS * NKEY, NKEY, NDH, nullptr);

  // 3) per-row top-32 of 256
  topk_256_32<<<(NZ * NROWS) / 8, 256, 0, stream>>>(dots, scores, sidx);

  // 4) combine halves, final top-32, softmax, gather values -> ctx16
  combine_gather<<<(NB * NH * NT) / 8, 256, 0, stream>>>(scores, sidx, values, ctx16);

  // 5) out = ctx @ Wout^T + bias
  gemm_wmma<2><<<dim3(NDIM / 128, NROWS / 64, 1), 256, 0, stream>>>(
      ctx16, 0LL, NDIM, wout16, 0LL, NDIM,
      (void*)out, 0LL, NDIM, NDIM, bias);
}
